// MemoryAugmentedAttention_83442624627082
// MI455X (gfx1250) — matrix operand order repaired and checked by reading, not compiled, not run on hardware
//
#include <hip/hip_runtime.h>
#include <hip/hip_bf16.h>
#include <stdint.h>

typedef __attribute__((ext_vector_type(16))) __bf16 v16bf;
typedef __attribute__((ext_vector_type(8)))  float  v8f;
typedef __attribute__((ext_vector_type(4)))  unsigned int ui4;
typedef __attribute__((ext_vector_type(8)))  int  i8v;
typedef __attribute__((ext_vector_type(4)))  int  i4v;

#define B_   2
#define T_   2048
#define D_   1024
#define H_   16
#define S_   64
#define DH_  64
#define MTOK (B_*T_)                   /* 4096 */
#define HEADSZ ((size_t)T_*DH_)        /* elements per (b,h) plane */
#define QKVSEG ((size_t)B_*H_*T_*DH_)  /* elements per Q/K/V segment */

#if __has_builtin(__builtin_amdgcn_tensor_load_to_lds)
#define HAVE_TDM 1
#else
#define HAVE_TDM 0
#endif

// ---------- helpers ----------

__device__ __forceinline__ __bf16 f2bf(float f) {
  union { float f; uint32_t u; } x; x.f = f;
  uint32_t r = x.u + 0x7fffu + ((x.u >> 16) & 1u);   // round-to-nearest-even
  union { uint16_t s; __bf16 b; } y; y.s = (uint16_t)(r >> 16);
  return y.b;
}

union Frag16 { v16bf v; uint4 q[2]; };

// A fragment (16x32, MxK): lane holds row m=lane%16; element e -> k = hi*16 + e
__device__ __forceinline__ v16bf load_frag_a(const __bf16* base, int ld, int m, int kc, int lane) {
  int hi = lane >> 4;
  Frag16 u;
  u.q[0] = *(const uint4*)(base + (size_t)m * ld + kc + hi * 16);
  u.q[1] = *(const uint4*)(base + (size_t)m * ld + kc + hi * 16 + 8);
  return u.v;
}

// B fragment (32x16, KxN): lane holds col n=lane%16; element e -> k = hi*16 + e
__device__ __forceinline__ v16bf load_frag_b(const __bf16* base, int ld, int n, int kc, int lane) {
  int hi = lane >> 4;
  Frag16 u;
  const uint4* p = (const uint4*)(base + (size_t)n * ld + kc + hi * 16);
  u.q[0] = p[0]; u.q[1] = p[1];
  return u.v;
}

__device__ __forceinline__ v8f wmma_bf16(v16bf a, v16bf b, v8f c) {
  return __builtin_amdgcn_wmma_f32_16x16x32_bf16(false, a, false, b, (short)0, c, false, false);
}

// gfx1250 async memory->LDS copy (ASYNCcnt-tracked, no VGPR data path)
__device__ __forceinline__ void async_ld_b128(__bf16* lds, const __bf16* g) {
  uint32_t l = (uint32_t)(uintptr_t)lds;   // LDS aperture: low 32 bits = LDS byte address
  asm volatile("global_load_async_to_lds_b128 %0, %1, off"
               :: "v"(l), "v"(g) : "memory");
}
__device__ __forceinline__ void wait_async0() {
  asm volatile("s_wait_asynccnt 0" ::: "memory");
}

#if HAVE_TDM
// TDM: one instruction moves a 32x64 bf16 tile (rows of a (T,64) bf16 tensor) into LDS.
// D# per CDNA5 ISA (2D, data_size=2B, type=2).
// Toolchain here is the 6-arg builtin: (g0 u32x4, g1 i32x8, g2 i32x4, g3 i32x4, i32x8, cpol)
__device__ __forceinline__ void tdm_load_tile32x64(const __bf16* gtile, __bf16* lds) {
  unsigned long long ga = (unsigned long long)(uintptr_t)gtile;
  ui4 g0;
  g0[0] = 1u;                                             // count=1, no gather
  g0[1] = (unsigned)(uintptr_t)lds;                       // lds_addr
  g0[2] = (unsigned)(ga & 0xffffffffu);                   // global_addr[31:0]
  g0[3] = (unsigned)((ga >> 32) & 0x01ffffffu) | (2u << 30); // global_addr[56:32] | type=2
  i8v g1;
  g1[0] = 0x00010000;            // workgroup_mask=0, data_size=1 (2 bytes)
  g1[1] = (int)(64u << 16);      // tensor_dim0 = 64 (low 16 in [31:16])
  g1[2] = (int)(2048u << 16);    // tensor_dim0 hi=0 | tensor_dim1 lo16 = T
  g1[3] = (int)(64u << 16);      // tensor_dim1 hi=0 | tile_dim0 = 64
  g1[4] = 32;                    // tile_dim1 = 32, tile_dim2 = 0
  g1[5] = 64;                    // tensor_dim0_stride = 64
  g1[6] = 0;                     // stride hi / tensor_dim1_stride lo
  g1[7] = 0;
  i4v z4 = {0, 0, 0, 0};
  i8v z8 = {0, 0, 0, 0, 0, 0, 0, 0};
  __builtin_amdgcn_tensor_load_to_lds(g0, g1, z4, z4, z8, 0);
}
#endif

// ---------- conversion kernels ----------

__global__ void cvt_f32_bf16(const float* __restrict__ src, __bf16* __restrict__ dst, int n) {
  int i = blockIdx.x * 256 + threadIdx.x;
  if (i < n) dst[i] = f2bf(src[i]);
}

__global__ void cvt_transpose_mv(const float* __restrict__ mv, __bf16* __restrict__ mvt) {
  int i = blockIdx.x * 256 + threadIdx.x;   // over D_*S_
  int d = i >> 6, s = i & (S_ - 1);
  mvt[(size_t)d * S_ + s] = f2bf(mv[(size_t)s * D_ + d]);
}

// ---------- generic bf16 WMMA GEMM: C(MxN) = A(MxK) * W(NxK)^T + bias ----------
// Tile 128x64, 4 waves; each wave: 32-row strip (2 A frags), 8 WMMA per 32-k step.
// MODE 0: write f32 C (ldc = N).  MODE 1: QKV epilogue -> bf16 scatter into fused QKV buffer.

template <int MODE>
__global__ __launch_bounds__(128) void gemm_bf16_kernel(
    const __bf16* __restrict__ A, const __bf16* __restrict__ W,
    const float* __restrict__ bias, float* __restrict__ C,
    __bf16* __restrict__ QKVb, int M, int N, int K) {
  __shared__ __align__(16) __bf16 As[128 * 32];
  __shared__ __align__(16) __bf16 Bs[64 * 32];
  const int tid = threadIdx.x, lane = tid & 31, wave = tid >> 5;
  const int hi = lane >> 4, nl = lane & 15;
  const int m0 = blockIdx.y * 128, n0 = blockIdx.x * 64;
  v8f acc0[4] = {}, acc1[4] = {};
  for (int kt = 0; kt < K; kt += 32) {
    // async global->LDS tile copies: A 128x32 (512 x b128), B 64x32 (256 x b128)
#pragma unroll
    for (int i = 0; i < 6; ++i) {
      int idx = i * 128 + tid;
      if (idx < 512) {
        int row = idx >> 2, seg = idx & 3;
        async_ld_b128(As + row * 32 + seg * 8,
                      A + (size_t)(m0 + row) * K + kt + seg * 8);
      } else {
        int j = idx - 512;
        int row = j >> 2, seg = j & 3;
        async_ld_b128(Bs + row * 32 + seg * 8,
                      W + (size_t)(n0 + row) * K + kt + seg * 8);
      }
    }
    wait_async0();
    __syncthreads();
    v16bf a0 = load_frag_a(As, 32, wave * 32 + nl, 0, lane);
    v16bf a1 = load_frag_a(As, 32, wave * 32 + 16 + nl, 0, lane);
#pragma unroll
    for (int nt = 0; nt < 4; ++nt) {
      v16bf b = load_frag_b(Bs, 32, nt * 16 + nl, 0, lane);
      acc0[nt] = wmma_bf16(a0, b, acc0[nt]);
      acc1[nt] = wmma_bf16(a1, b, acc1[nt]);
    }
    __syncthreads();
  }
#pragma unroll
  for (int blk = 0; blk < 2; ++blk) {
#pragma unroll
    for (int nt = 0; nt < 4; ++nt) {
#pragma unroll
      for (int v = 0; v < 8; ++v) {
        int m = m0 + wave * 32 + blk * 16 + hi * 8 + v;
        int n = n0 + nt * 16 + nl;
        float val = (blk ? acc1[nt][v] : acc0[nt][v]) + (bias ? bias[n] : 0.f);
        if (MODE == 0) {
          C[(size_t)m * N + n] = val;
        } else {
          int b = m >> 11, t = m & (T_ - 1);
          int which = n >> 10, h = (n >> 6) & (H_ - 1), dd = n & (DH_ - 1);
          QKVb[(size_t)which * QKVSEG + (((size_t)b * H_ + h) * T_ + t) * DH_ + dd] = f2bf(val);
        }
      }
    }
  }
}

// ---------- flash attention (causal), 1 WG = (b,h, 64-row q tile), 4 waves ----------
// K/V tiles fetched by the Tensor Data Mover (double-buffered, TENSORcnt-paced);
// falls back to per-lane async-to-LDS if the TDM builtin is unavailable.

__global__ __launch_bounds__(128) void flash_attn_kernel(
    const __bf16* __restrict__ Qg, const __bf16* __restrict__ Kg,
    const __bf16* __restrict__ Vg, float* __restrict__ seq_out) {
  __shared__ __align__(16) __bf16 Ks[2][32 * 64];
  __shared__ __align__(16) __bf16 Vs[2][32 * 64];
  __shared__ __align__(16) __bf16 Ps[64 * 32];
  const int qt = blockIdx.x, bh = blockIdx.y;
  const int tid = threadIdx.x, lane = tid & 31, wave = tid >> 5;
  const int hi = lane >> 4, nl = lane & 15;
  const __bf16* Qh = Qg + (size_t)bh * HEADSZ;
  const __bf16* Kh = Kg + (size_t)bh * HEADSZ;
  const __bf16* Vh = Vg + (size_t)bh * HEADSZ;

  const int mrow = qt * 64 + wave * 16 + nl;
  v16bf qa0 = load_frag_a(Qh, DH_, mrow, 0, lane);
  v16bf qa1 = load_frag_a(Qh, DH_, mrow, 32, lane);

  v8f o[4] = {};
  float mrun[8], lrun[8];
#pragma unroll
  for (int v = 0; v < 8; ++v) { mrun[v] = -1e30f; lrun[v] = 0.f; }
  const float scale = 0.125f;                 // dh^-0.5
  const int ktmax = 2 * qt + 1;               // 32-wide key tiles up to the diagonal
  int cur = 0;

#if HAVE_TDM
  if (wave == 0) {                            // TDM issue is per-wave; EXEC ignored
    tdm_load_tile32x64(Kh, Ks[0]);
    tdm_load_tile32x64(Vh, Vs[0]);
  }
#endif

  for (int kt = 0; kt <= ktmax; ++kt) {
#if HAVE_TDM
    if (wave == 0) {
      if (kt < ktmax) {                       // prefetch next tile into the other buffer
        tdm_load_tile32x64(Kh + (size_t)(kt + 1) * 32 * DH_, Ks[cur ^ 1]);
        tdm_load_tile32x64(Vh + (size_t)(kt + 1) * 32 * DH_, Vs[cur ^ 1]);
        __builtin_amdgcn_s_wait_tensorcnt(2); // tile kt's 2 loads (in-order) done
      } else {
        __builtin_amdgcn_s_wait_tensorcnt(0);
      }
    }
#else
#pragma unroll
    for (int i = 0; i < 2; ++i) {
      int idx = i * 128 + tid;
      int row = idx >> 3, seg = idx & 7;
      async_ld_b128(Ks[cur] + row * DH_ + seg * 8,
                    Kh + ((size_t)kt * 32 + row) * DH_ + seg * 8);
      async_ld_b128(Vs[cur] + row * DH_ + seg * 8,
                    Vh + ((size_t)kt * 32 + row) * DH_ + seg * 8);
    }
    wait_async0();
#endif
    __syncthreads();
    const __bf16* Kt = Ks[cur];
    const __bf16* Vt = Vs[cur];

    // S strip (16x32) = Q strip (16x64) x K_tile^T
    v8f s0 = {}, s1 = {};
    {
      v16bf b0 = load_frag_b(Kt, DH_, nl, 0, lane);
      v16bf b1 = load_frag_b(Kt, DH_, nl, 32, lane);
      s0 = wmma_bf16(qa0, b0, s0);
      s0 = wmma_bf16(qa1, b1, s0);
      v16bf c0 = load_frag_b(Kt, DH_, 16 + nl, 0, lane);
      v16bf c1 = load_frag_b(Kt, DH_, 16 + nl, 32, lane);
      s1 = wmma_bf16(qa0, c0, s1);
      s1 = wmma_bf16(qa1, c1, s1);
    }

    // causal mask + online softmax (row stats redundant across the 16 n-lanes)
#pragma unroll
    for (int v = 0; v < 8; ++v) {
      int qg = qt * 64 + wave * 16 + hi * 8 + v;
      int j0 = kt * 32 + nl, j1 = kt * 32 + 16 + nl;
      float x0 = (j0 <= qg) ? s0[v] * scale : -1e30f;
      float x1 = (j1 <= qg) ? s1[v] * scale : -1e30f;
      float rm = fmaxf(x0, x1);
      rm = fmaxf(rm, __shfl_xor(rm, 1, 32));
      rm = fmaxf(rm, __shfl_xor(rm, 2, 32));
      rm = fmaxf(rm, __shfl_xor(rm, 4, 32));
      rm = fmaxf(rm, __shfl_xor(rm, 8, 32));
      float mnew = fmaxf(mrun[v], rm);
      float corr = __expf(mrun[v] - mnew);
      mrun[v] = mnew;
      float p0 = __expf(x0 - mnew), p1 = __expf(x1 - mnew);
      float rs = p0 + p1;
      rs += __shfl_xor(rs, 1, 32);
      rs += __shfl_xor(rs, 2, 32);
      rs += __shfl_xor(rs, 4, 32);
      rs += __shfl_xor(rs, 8, 32);
      lrun[v] = lrun[v] * corr + rs;
#pragma unroll
      for (int nt = 0; nt < 4; ++nt) o[nt][v] *= corr;
      int prow = wave * 16 + hi * 8 + v;        // P tile: C-layout -> A-layout via LDS
      Ps[prow * 32 + nl]      = f2bf(p0);
      Ps[prow * 32 + 16 + nl] = f2bf(p1);
    }
    __syncthreads();

    // O strip (16x64) += P (16x32) x V_tile (32x64)
    v16bf pa = load_frag_a(Ps, 32, wave * 16 + nl, 0, lane);
#pragma unroll
    for (int nt = 0; nt < 4; ++nt) {
      v16bf vb = {};
#pragma unroll
      for (int e = 0; e < 16; ++e)
        vb[e] = Vt[(hi * 16 + e) * DH_ + nt * 16 + nl];
      o[nt] = wmma_bf16(pa, vb, o[nt]);
    }
    __syncthreads();   // all reads of buffer `cur` done before it is refilled
    cur ^= 1;
  }

  const int b = bh >> 4, h = bh & 15;
#pragma unroll
  for (int nt = 0; nt < 4; ++nt) {
#pragma unroll
    for (int v = 0; v < 8; ++v) {
      int t = qt * 64 + wave * 16 + hi * 8 + v;
      int dd = nt * 16 + nl;
      seq_out[((size_t)(b * T_ + t)) * D_ + h * DH_ + dd] = o[nt][v] / lrun[v];
    }
  }
}

// ---------- memory path: Plucker lines, quadratic scoring, softmax, mean over heads ----------

__global__ __launch_bounds__(64) void memscore_kernel(
    const float* __restrict__ P12, const float* __restrict__ grams,
    float* __restrict__ memATT) {
  __shared__ float G[S_][36];
  __shared__ float red[2];
  const int tok = blockIdx.x;
  const int s = threadIdx.x;
  for (int i = 0; i < 36; ++i) {
    int rr = i / 6, cc = i % 6;
    G[s][i] = 0.5f * (grams[s * 36 + rr * 6 + cc] + grams[s * 36 + cc * 6 + rr]);
  }
  __syncthreads();
  const float* p = P12 + (size_t)tok * 128;
  float acc = 0.f;
  for (int h = 0; h < H_; ++h) {
    float p1[4], p2[4];
#pragma unroll
    for (int c = 0; c < 4; ++c) { p1[c] = p[h * 4 + c]; p2[c] = p[64 + h * 4 + c]; }
    float L[6];
    L[0] = p1[0] * p2[1] - p1[1] * p2[0];
    L[1] = p1[0] * p2[2] - p1[2] * p2[0];
    L[2] = p1[0] * p2[3] - p1[3] * p2[0];
    L[3] = p1[1] * p2[2] - p1[2] * p2[1];
    L[4] = p1[1] * p2[3] - p1[3] * p2[1];
    L[5] = p1[2] * p2[3] - p1[3] * p2[2];
    float nrm = sqrtf(L[0]*L[0] + L[1]*L[1] + L[2]*L[2] + L[3]*L[3] + L[4]*L[4] + L[5]*L[5]);
    float inv = 1.f / fmaxf(nrm, 1e-12f);
    float Jl[6] = { L[5]*inv, -L[4]*inv, L[3]*inv, L[2]*inv, -L[1]*inv, L[0]*inv };
    float sc = 0.f;
#pragma unroll
    for (int i = 0; i < 6; ++i)
#pragma unroll
      for (int j = 0; j < 6; ++j)
        sc += Jl[i] * G[s][i * 6 + j] * Jl[j];
    float mx = sc;
    mx = fmaxf(mx, __shfl_xor(mx, 1, 32));
    mx = fmaxf(mx, __shfl_xor(mx, 2, 32));
    mx = fmaxf(mx, __shfl_xor(mx, 4, 32));
    mx = fmaxf(mx, __shfl_xor(mx, 8, 32));
    mx = fmaxf(mx, __shfl_xor(mx, 16, 32));
    if ((s & 31) == 0) red[s >> 5] = mx;
    __syncthreads();
    mx = fmaxf(red[0], red[1]);
    __syncthreads();
    float e = __expf(sc - mx);
    float sm = e;
    sm += __shfl_xor(sm, 1, 32);
    sm += __shfl_xor(sm, 2, 32);
    sm += __shfl_xor(sm, 4, 32);
    sm += __shfl_xor(sm, 8, 32);
    sm += __shfl_xor(sm, 16, 32);
    if ((s & 31) == 0) red[s >> 5] = sm;
    __syncthreads();
    sm = red[0] + red[1];
    __syncthreads();
    acc += e / sm;
  }
  memATT[(size_t)tok * S_ + s] = acc * (1.f / (float)H_);
}

// ---------- gate: mean_h sigmoid(x @ gate_w.T + gate_b) ----------

__global__ __launch_bounds__(256) void gate_kernel(
    const float* __restrict__ x, const float* __restrict__ gw,
    const float* __restrict__ gb, float* __restrict__ gate) {
  __shared__ float xs[D_];
  __shared__ float hsum[H_];
  const int tok = blockIdx.x, tid = threadIdx.x;
  for (int i = tid; i < D_; i += 256) xs[i] = x[(size_t)tok * D_ + i];
  __syncthreads();
  const int h = tid >> 4, li = tid & 15;
  float a = 0.f;
  for (int k = li; k < D_; k += 16) a += xs[k] * gw[(size_t)h * D_ + k];
  a += __shfl_xor(a, 1, 32);
  a += __shfl_xor(a, 2, 32);
  a += __shfl_xor(a, 4, 32);
  a += __shfl_xor(a, 8, 32);
  if (li == 0) hsum[h] = 1.f / (1.f + __expf(-(a + gb[h])));
  __syncthreads();
  if (tid == 0) {
    float g = 0.f;
    for (int i = 0; i < H_; ++i) g += hsum[i];
    gate[tok] = g / (float)H_;
  }
}

// ---------- combine: bf16(seq_out + gate * mem_read) ----------

__global__ void combine_kernel(const float* __restrict__ seq, const float* __restrict__ memread,
                               const float* __restrict__ gate, __bf16* __restrict__ comb) {
  size_t i = (size_t)blockIdx.x * 256 + threadIdx.x;
  int tok = (int)(i >> 10);
  comb[i] = f2bf(seq[i] + gate[tok] * memread[i]);
}

// ---------- host launcher ----------

extern "C" void kernel_launch(void* const* d_in, const int* in_sizes, int n_in,
                              void* d_out, int out_size, void* d_ws, size_t ws_size,
                              hipStream_t stream) {
  const float* x          = (const float*)d_in[0];
  const float* qkv_w      = (const float*)d_in[1];
  const float* qkv_b      = (const float*)d_in[2];
  const float* w1         = (const float*)d_in[3];
  const float* w2         = (const float*)d_in[4];
  const float* mem_grams  = (const float*)d_in[5];
  const float* mem_values = (const float*)d_in[6];
  const float* gate_w     = (const float*)d_in[7];
  const float* gate_b     = (const float*)d_in[8];
  const float* out_w      = (const float*)d_in[9];
  const float* out_b      = (const float*)d_in[10];
  float* out = (float*)d_out;

  char* ws = (char*)d_ws;
  size_t off = 0;
  auto alloc = [&](size_t bytes) -> void* {
    void* p = ws + off;
    off = (off + bytes + 255) & ~(size_t)255;
    return p;
  };
  __bf16* xh      = (__bf16*)alloc((size_t)MTOK * D_ * 2);
  __bf16* wqkvh   = (__bf16*)alloc((size_t)3 * D_ * D_ * 2);
  __bf16* pwh     = (__bf16*)alloc((size_t)128 * D_ * 2);
  __bf16* outwh   = (__bf16*)alloc((size_t)D_ * D_ * 2);
  __bf16* mvth    = (__bf16*)alloc((size_t)D_ * S_ * 2);
  __bf16* qkvb    = (__bf16*)alloc(3 * QKVSEG * 2);
  float*  seqo    = (float*)alloc((size_t)MTOK * D_ * 4);
  float*  Pbuf    = (float*)alloc((size_t)MTOK * 128 * 4);
  float*  memATT  = (float*)alloc((size_t)MTOK * S_ * 4);
  __bf16* memATTh = (__bf16*)alloc((size_t)MTOK * S_ * 2);
  float*  memread = (float*)alloc((size_t)MTOK * D_ * 4);
  float*  gateb   = (float*)alloc((size_t)MTOK * 4);
  __bf16* combh   = (__bf16*)alloc((size_t)MTOK * D_ * 2);

  // conversions
  cvt_f32_bf16<<<(MTOK * D_) / 256, 256, 0, stream>>>(x, xh, MTOK * D_);
  cvt_f32_bf16<<<(3 * D_ * D_) / 256, 256, 0, stream>>>(qkv_w, wqkvh, 3 * D_ * D_);
  cvt_f32_bf16<<<(64 * D_) / 256, 256, 0, stream>>>(w1, pwh, 64 * D_);
  cvt_f32_bf16<<<(64 * D_) / 256, 256, 0, stream>>>(w2, pwh + 64 * D_, 64 * D_);
  cvt_f32_bf16<<<(D_ * D_) / 256, 256, 0, stream>>>(out_w, outwh, D_ * D_);
  cvt_transpose_mv<<<(D_ * S_) / 256, 256, 0, stream>>>(mem_values, mvth);

  // QKV projection (scatter epilogue into fused bf16 QKV buffer)
  {
    dim3 g(3 * D_ / 64, MTOK / 128);
    gemm_bf16_kernel<1><<<g, 128, 0, stream>>>(xh, wqkvh, qkv_b, nullptr, qkvb,
                                               MTOK, 3 * D_, D_);
  }
  // p1/p2 projection (N=128)
  {
    dim3 g(128 / 64, MTOK / 128);
    gemm_bf16_kernel<0><<<g, 128, 0, stream>>>(xh, pwh, nullptr, Pbuf, nullptr,
                                               MTOK, 128, D_);
  }
  // causal flash attention
  {
    dim3 g(T_ / 64, B_ * H_);
    flash_attn_kernel<<<g, 128, 0, stream>>>(qkvb, qkvb + QKVSEG, qkvb + 2 * QKVSEG, seqo);
  }
  // memory scoring + softmax + head mean
  memscore_kernel<<<MTOK, 64, 0, stream>>>(Pbuf, mem_grams, memATT);
  cvt_f32_bf16<<<(MTOK * S_) / 256, 256, 0, stream>>>(memATT, memATTh, MTOK * S_);
  // mem_read = memATT @ mem_values  (K=64)
  {
    dim3 g(D_ / 64, MTOK / 128);
    gemm_bf16_kernel<0><<<g, 128, 0, stream>>>(memATTh, mvth, nullptr, memread, nullptr,
                                               MTOK, D_, S_);
  }
  // gate
  gate_kernel<<<MTOK, 256, 0, stream>>>(x, gate_w, gate_b, gateb);
  // combined = seq_out + gate * mem_read (bf16)
  combine_kernel<<<(MTOK * D_) / 256, 256, 0, stream>>>(seqo, memread, gateb, combh);
  // out = combined @ out_w.T + out_b
  {
    dim3 g(D_ / 64, MTOK / 128);
    gemm_bf16_kernel<0><<<g, 128, 0, stream>>>(combh, outwh, out_b, out, nullptr,
                                               MTOK, D_, D_);
  }
}
